// SelfAttention_89704686944675
// MI455X (gfx1250) — matrix operand order repaired and checked by reading, not compiled, not run on hardware
//
#include <hip/hip_runtime.h>

// ----------------------------------------------------------------------------
// Self-attention for MI455X (gfx1250, wave32, WMMA, async-to-LDS).
//
//   K1 qkv_gemm   : Qb/Kb/Vb (bf16) = cast(X) @ cast(W*), Q pre-scaled 1/sqrt(d)
//   K2 score_gemm : S (f32)  = Qb @ Kb^T   (double-buffered async-to-LDS)
//   K3 softmax    : P (bf16) = softmax_rows(S)
//   K4 out_gemm   : O (f32)  = P @ Vb      (async A tile, VGPR-transposed V tile)
//
// Workspace layout (432 MB):
//   [0,16M) Qb | [16M,32M) Kb | [32M,48M) Vb   bf16 8192x1024 each
//   [48M,304M)  S f32 8192x8192
//   [304M,432M) P bf16 8192x8192
// ----------------------------------------------------------------------------

typedef __attribute__((ext_vector_type(16))) __bf16        v16bf;
typedef __attribute__((ext_vector_type(8)))  float         v8f;
typedef __attribute__((ext_vector_type(4))) unsigned int   u32x4;
typedef __attribute__((ext_vector_type(2))) unsigned int   u32x2;
typedef __attribute__((ext_vector_type(4)))  float         f32x4;
typedef int i32x4v __attribute__((vector_size(16)));       // matches builtin V4i

#define NROW 8192
#define DIM  1024
#define BT   128      // block tile (M and N)
#define KT   32       // k-step (bf16 WMMA 16x16x32)
#define LDST 40       // LDS row stride in shorts (80B: 16B-aligned, bank-spread)

#ifndef __has_builtin
#define __has_builtin(x) 0
#endif

#define GLOBAL_AS __attribute__((address_space(1)))
#define LDS_AS    __attribute__((address_space(3)))

// ---- CDNA5 async global->LDS copy (ASYNCcnt-tracked) -----------------------
static __device__ __forceinline__ void async_copy_b128(void* lds, const void* g) {
#if __has_builtin(__builtin_amdgcn_global_load_async_to_lds_b128)
  __builtin_amdgcn_global_load_async_to_lds_b128(
      (GLOBAL_AS i32x4v*)g, (LDS_AS i32x4v*)lds, 0, 0);
#else
  // LDS aperture: low 32 bits of a generic shared pointer are the LDS address.
  unsigned lo = (unsigned)(unsigned long long)lds;
  asm volatile("global_load_async_to_lds_b128 %0, %1, off"
               :: "v"(lo), "v"(g) : "memory");
#endif
}

static __device__ __forceinline__ void wait_async0() {
#if __has_builtin(__builtin_amdgcn_s_wait_asynccnt)
  __builtin_amdgcn_s_wait_asynccnt(0);
#else
  asm volatile("s_wait_asynccnt 0x0" ::: "memory");
#endif
}

// ---- helpers ---------------------------------------------------------------
static __device__ __forceinline__ unsigned short f2bf(float f) {
  unsigned int u = __float_as_uint(f);
  unsigned int r = u + 0x7FFFu + ((u >> 16) & 1u);   // round-to-nearest-even
  return (unsigned short)(r >> 16);
}
static __device__ __forceinline__ unsigned int pack2bf(float a, float b) {
  return (unsigned int)f2bf(a) | ((unsigned int)f2bf(b) << 16);
}

// A fragment: 16x32 bf16 (ISA 7.12.2): lane half selects K sub-ranges.
static __device__ __forceinline__ v16bf load_frag_a(const unsigned short* lds,
                                                    int mBase, int lane) {
  const int m  = mBase + (lane & 15);
  const int ko = (lane >> 4) << 3;               // 0 or 8 shorts
  const unsigned short* p = lds + m * LDST + ko;
  union { v16bf bf; u32x4 q[2]; } f;
  f.q[0] = *(const u32x4*)(p);                   // K = ko+0..7
  f.q[1] = *(const u32x4*)(p + 16);              // K = 16+ko+0..7
  return f.bf;
}

// B fragment: 32x16 bf16, tile stored [n][k] in LDS; lane half selects K sub-ranges.
static __device__ __forceinline__ v16bf load_frag_b(const unsigned short* lds,
                                                    int nBase, int lane) {
  const int n  = nBase + (lane & 15);
  const int ko = (lane >> 4) << 3;               // 0 or 8 shorts
  const unsigned short* p = lds + n * LDST + ko;
  union { v16bf bf; u32x4 q[2]; } f;
  f.q[0] = *(const u32x4*)(p);
  f.q[1] = *(const u32x4*)(p + 16);
  return f.bf;
}

static __device__ __forceinline__ v8f wmma_bf16(v16bf a, v16bf b, v8f c) {
  return __builtin_amdgcn_wmma_f32_16x16x32_bf16(false, a, false, b,
                                                 (short)0, c, false, false);
}

// ---------------------------------------------------------------------------
// K1: Qb/Kb/Vb = cast_bf16(X @ W).  blockIdx.z selects {Q,K,V}.
// (f32->bf16 conversion in the staging path; single-buffered.)
// ---------------------------------------------------------------------------
__global__ __launch_bounds__(256)
void qkv_gemm_kernel(const float* __restrict__ X,
                     const float* __restrict__ WQ,
                     const float* __restrict__ WK,
                     const float* __restrict__ WV,
                     unsigned short* __restrict__ Qb,
                     unsigned short* __restrict__ Kb,
                     unsigned short* __restrict__ Vb) {
  __shared__ unsigned short lsA[BT * LDST];
  __shared__ unsigned short lsB[BT * LDST];

  const int t = threadIdx.x, lane = t & 31, wave = t >> 5;
  const int wm = wave >> 2, wn = wave & 3;                  // 2x4 wave grid
  const int m0 = blockIdx.y * BT, n0 = blockIdx.x * BT;

  const float* W; unsigned short* out; float scale;
  if (blockIdx.z == 0)      { W = WQ; out = Qb; scale = 0.03125f; } // 1/sqrt(1024)
  else if (blockIdx.z == 1) { W = WK; out = Kb; scale = 1.0f; }
  else                      { W = WV; out = Vb; scale = 1.0f; }

  const v8f zero = {0.f,0.f,0.f,0.f,0.f,0.f,0.f,0.f};
  v8f acc[4][2];
  #pragma unroll
  for (int i = 0; i < 4; ++i)
    #pragma unroll
    for (int j = 0; j < 2; ++j) acc[i][j] = zero;

  const int ar = t >> 3, ac = (t & 7) << 2;
  const int bn = t & 127, bk = t >> 7;

  for (int k0 = 0; k0 < DIM; k0 += KT) {
    #pragma unroll
    for (int rr = 0; rr < BT; rr += 32) {
      f32x4 v = *(const f32x4*)(X + (size_t)(m0 + ar + rr) * DIM + k0 + ac);
      u32x2 pk; pk[0] = pack2bf(v[0], v[1]); pk[1] = pack2bf(v[2], v[3]);
      *(u32x2*)(&lsA[(ar + rr) * LDST + ac]) = pk;
    }
    #pragma unroll
    for (int kk = 0; kk < KT; kk += 2) {
      float wv = W[(size_t)(k0 + bk + kk) * DIM + n0 + bn];
      lsB[bn * LDST + bk + kk] = f2bf(wv);
    }
    if (k0 + KT < DIM)
      __builtin_prefetch(X + (size_t)(m0 + ar) * DIM + k0 + KT + ac, 0, 1);
    __syncthreads();

    v16bf af[4], bb[2];
    #pragma unroll
    for (int mt = 0; mt < 4; ++mt) af[mt] = load_frag_a(lsA, wm * 64 + mt * 16, lane);
    #pragma unroll
    for (int nt = 0; nt < 2; ++nt) bb[nt] = load_frag_b(lsB, wn * 32 + nt * 16, lane);
    #pragma unroll
    for (int mt = 0; mt < 4; ++mt)
      #pragma unroll
      for (int nt = 0; nt < 2; ++nt)
        acc[mt][nt] = wmma_bf16(af[mt], bb[nt], acc[mt][nt]);
    __syncthreads();
  }

  const int rofs = (lane >> 4) << 3, cofs = lane & 15;
  #pragma unroll
  for (int mt = 0; mt < 4; ++mt)
    #pragma unroll
    for (int nt = 0; nt < 2; ++nt)
      #pragma unroll
      for (int r = 0; r < 8; ++r) {
        const int gm = m0 + wm * 64 + mt * 16 + r + rofs;
        const int gn = n0 + wn * 32 + nt * 16 + cofs;
        out[(size_t)gm * DIM + gn] = f2bf(acc[mt][nt][r] * scale);
      }
}

// ---------------------------------------------------------------------------
// K2: S = Qb @ Kb^T.  Double-buffered, fully async-to-LDS staged.
// Schedule per k-step: wait_async; barrier; issue copy(i+1); compute(i).
// ---------------------------------------------------------------------------
__global__ __launch_bounds__(256)
void score_gemm_kernel(const unsigned short* __restrict__ Qb,
                       const unsigned short* __restrict__ Kb,
                       float* __restrict__ S) {
  __shared__ unsigned short lsA[2][BT * LDST];
  __shared__ unsigned short lsB[2][BT * LDST];

  const int t = threadIdx.x, lane = t & 31, wave = t >> 5;
  const int wm = wave >> 2, wn = wave & 3;
  const int m0 = blockIdx.y * BT, n0 = blockIdx.x * BT;

  const v8f zero = {0.f,0.f,0.f,0.f,0.f,0.f,0.f,0.f};
  v8f acc[4][2];
  #pragma unroll
  for (int i = 0; i < 4; ++i)
    #pragma unroll
    for (int j = 0; j < 2; ++j) acc[i][j] = zero;

  const int ar = t >> 2, ac8 = (t & 3) << 3;   // 4 threads/row, 16B each

  auto issue = [&](int k0, int buf) {
    #pragma unroll
    for (int rr = 0; rr < BT; rr += 64) {
      async_copy_b128(&lsA[buf][(ar + rr) * LDST + ac8],
                      Qb + (size_t)(m0 + ar + rr) * DIM + k0 + ac8);
      async_copy_b128(&lsB[buf][(ar + rr) * LDST + ac8],
                      Kb + (size_t)(n0 + ar + rr) * DIM + k0 + ac8);
    }
  };

  issue(0, 0);
  int buf = 0;
  for (int k0 = 0; k0 < DIM; k0 += KT, buf ^= 1) {
    wait_async0();          // my copies of tile i complete
    __syncthreads();        // everyone's copies visible; compute(i-1) done
    if (k0 + KT < DIM) issue(k0 + KT, buf ^ 1);   // overlaps compute(i)

    v16bf af[4], bb[2];
    #pragma unroll
    for (int mt = 0; mt < 4; ++mt) af[mt] = load_frag_a(lsA[buf], wm * 64 + mt * 16, lane);
    #pragma unroll
    for (int nt = 0; nt < 2; ++nt) bb[nt] = load_frag_b(lsB[buf], wn * 32 + nt * 16, lane);
    #pragma unroll
    for (int mt = 0; mt < 4; ++mt)
      #pragma unroll
      for (int nt = 0; nt < 2; ++nt)
        acc[mt][nt] = wmma_bf16(af[mt], bb[nt], acc[mt][nt]);
  }

  const int rofs = (lane >> 4) << 3, cofs = lane & 15;
  #pragma unroll
  for (int mt = 0; mt < 4; ++mt)
    #pragma unroll
    for (int nt = 0; nt < 2; ++nt)
      #pragma unroll
      for (int r = 0; r < 8; ++r) {
        const int gm = m0 + wm * 64 + mt * 16 + r + rofs;
        const int gn = n0 + wn * 32 + nt * 16 + cofs;
        S[(size_t)gm * NROW + gn] = acc[mt][nt][r];
      }
}

// ---------------------------------------------------------------------------
// K3: row-wise softmax, one block per row; 32 f32 per thread held in regs.
// ---------------------------------------------------------------------------
__global__ __launch_bounds__(256)
void softmax_kernel(const float* __restrict__ S, unsigned short* __restrict__ P) {
  __shared__ float red[256];
  const int t = threadIdx.x;
  const size_t row = blockIdx.x;
  const f32x4* rs = (const f32x4*)(S + row * NROW);

  float x[32];
  float lmax = -3.402823466e38f;
  #pragma unroll
  for (int i = 0; i < 8; ++i) {
    f32x4 v = rs[t + (i << 8)];
    #pragma unroll
    for (int j = 0; j < 4; ++j) { x[i * 4 + j] = v[j]; lmax = fmaxf(lmax, v[j]); }
  }
  red[t] = lmax; __syncthreads();
  #pragma unroll
  for (int s = 128; s > 0; s >>= 1) {
    if (t < s) red[t] = fmaxf(red[t], red[t + s]);
    __syncthreads();
  }
  const float mx = red[0]; __syncthreads();

  float lsum = 0.f;
  #pragma unroll
  for (int i = 0; i < 32; ++i) { x[i] = __expf(x[i] - mx); lsum += x[i]; }
  red[t] = lsum; __syncthreads();
  #pragma unroll
  for (int s = 128; s > 0; s >>= 1) {
    if (t < s) red[t] = red[t] + red[t + s];
    __syncthreads();
  }
  const float inv = 1.0f / red[0];

  u32x2* pr = (u32x2*)(P + row * NROW);
  #pragma unroll
  for (int i = 0; i < 8; ++i) {
    u32x2 pk;
    pk[0] = pack2bf(x[i * 4 + 0] * inv, x[i * 4 + 1] * inv);
    pk[1] = pack2bf(x[i * 4 + 2] * inv, x[i * 4 + 3] * inv);
    pr[t + (i << 8)] = pk;
  }
}

// ---------------------------------------------------------------------------
// K4: O = P @ Vb.  Double-buffered: async A tile, VGPR-transposed V tile.
// ---------------------------------------------------------------------------
__global__ __launch_bounds__(256)
void out_gemm_kernel(const unsigned short* __restrict__ P,
                     const unsigned short* __restrict__ Vb,
                     float* __restrict__ O) {
  __shared__ unsigned short lsA[2][BT * LDST];
  __shared__ unsigned short lsB[2][BT * LDST];

  const int t = threadIdx.x, lane = t & 31, wave = t >> 5;
  const int wm = wave >> 2, wn = wave & 3;
  const int m0 = blockIdx.y * BT, n0 = blockIdx.x * BT;

  const v8f zero = {0.f,0.f,0.f,0.f,0.f,0.f,0.f,0.f};
  v8f acc[4][2];
  #pragma unroll
  for (int i = 0; i < 4; ++i)
    #pragma unroll
    for (int j = 0; j < 2; ++j) acc[i][j] = zero;

  const int ar = t >> 2, ac8 = (t & 3) << 3;
  const int bn = t & 127, bk = t >> 7;

  auto issueA = [&](int k0, int buf) {
    #pragma unroll
    for (int rr = 0; rr < BT; rr += 64)
      async_copy_b128(&lsA[buf][(ar + rr) * LDST + ac8],
                      P + (size_t)(m0 + ar + rr) * NROW + k0 + ac8);
  };
  auto stageB = [&](int k0, int buf) {       // transpose Vb[k][n] -> LDS [n][k]
    #pragma unroll
    for (int kk = 0; kk < KT; kk += 2)
      lsB[buf][bn * LDST + bk + kk] = Vb[(size_t)(k0 + bk + kk) * DIM + n0 + bn];
  };

  issueA(0, 0);
  stageB(0, 0);
  int buf = 0;
  for (int k0 = 0; k0 < NROW; k0 += KT, buf ^= 1) {
    wait_async0();
    __syncthreads();
    if (k0 + KT < NROW) {
      issueA(k0 + KT, buf ^ 1);
      __builtin_prefetch(Vb + (size_t)(k0 + KT + bk) * DIM + n0 + bn, 0, 1);
      stageB(k0 + KT, buf ^ 1);
    }

    v16bf af[4], bb[2];
    #pragma unroll
    for (int mt = 0; mt < 4; ++mt) af[mt] = load_frag_a(lsA[buf], wm * 64 + mt * 16, lane);
    #pragma unroll
    for (int nt = 0; nt < 2; ++nt) bb[nt] = load_frag_b(lsB[buf], wn * 32 + nt * 16, lane);
    #pragma unroll
    for (int mt = 0; mt < 4; ++mt)
      #pragma unroll
      for (int nt = 0; nt < 2; ++nt)
        acc[mt][nt] = wmma_bf16(af[mt], bb[nt], acc[mt][nt]);
  }

  const int rofs = (lane >> 4) << 3, cofs = lane & 15;
  #pragma unroll
  for (int mt = 0; mt < 4; ++mt)
    #pragma unroll
    for (int nt = 0; nt < 2; ++nt)
      #pragma unroll
      for (int r = 0; r < 8; ++r) {
        const int gm = m0 + wm * 64 + mt * 16 + r + rofs;
        const int gn = n0 + wn * 32 + nt * 16 + cofs;
        O[(size_t)gm * DIM + gn] = acc[mt][nt][r];
      }
}

// ---------------------------------------------------------------------------
extern "C" void kernel_launch(void* const* d_in, const int* in_sizes, int n_in,
                              void* d_out, int out_size, void* d_ws, size_t ws_size,
                              hipStream_t stream) {
  (void)in_sizes; (void)n_in; (void)out_size; (void)ws_size;
  const float* X  = (const float*)d_in[0];
  const float* WQ = (const float*)d_in[1];
  const float* WK = (const float*)d_in[2];
  const float* WV = (const float*)d_in[3];
  float* O = (float*)d_out;

  const size_t nd = (size_t)NROW * DIM;
  unsigned short* Qb = (unsigned short*)d_ws;
  unsigned short* Kb = Qb + nd;
  unsigned short* Vb = Kb + nd;
  float*          S  = (float*)(Vb + nd);
  unsigned short* P  = (unsigned short*)(S + (size_t)NROW * NROW);

  qkv_gemm_kernel<<<dim3(DIM / BT, NROW / BT, 3), 256, 0, stream>>>(
      X, WQ, WK, WV, Qb, Kb, Vb);
  score_gemm_kernel<<<dim3(NROW / BT, NROW / BT), 256, 0, stream>>>(Qb, Kb, S);
  softmax_kernel<<<NROW, 256, 0, stream>>>(S, P);
  out_gemm_kernel<<<dim3(DIM / BT, NROW / BT), 256, 0, stream>>>(P, Vb, O);
}
